// MultiQueryAttention_2138893714153
// MI455X (gfx1250) — matrix operand order repaired and checked by reading, not compiled, not run on hardware
//
#include <hip/hip_runtime.h>
#include <stdint.h>

// ---------------- problem constants (from reference) ----------------
constexpr int Bb  = 2;
constexpr int S   = 2048;
constexpr int D   = 2048;
constexpr int H   = 16;
constexpr int DK  = 128;
constexpr int DV  = 128;
constexpr int BS  = Bb * S;            // 4096 rows
constexpr float SCALE = 0.08838834764831845f; // 1/sqrt(128)

// ---------------- WMMA / TDM types ----------------
typedef __attribute__((ext_vector_type(16))) __bf16 v16bf;
typedef __attribute__((ext_vector_type(8)))  float  v8f;
typedef __attribute__((ext_vector_type(4)))  unsigned int u32x4;
typedef __attribute__((ext_vector_type(8)))  int i32x8;
typedef __attribute__((ext_vector_type(4)))  int i32x4;

union Frag { v16bf v; uint32_t u[8]; };

__device__ __forceinline__ uint16_t f2bf(float f) {
    uint32_t u = __float_as_uint(f);
    u += 0x7FFFu + ((u >> 16) & 1u);   // round-to-nearest-even
    return (uint16_t)(u >> 16);
}

__device__ __forceinline__ v8f wmma_bf16(const Frag& a, const Frag& b, v8f c) {
    return __builtin_amdgcn_wmma_f32_16x16x32_bf16(
        false, a.v, false, b.v, (short)0, c, false, false);
}

// ---------------------------------------------------------------------
// Tensor Data Mover: 2D bf16 tile (tile_d1 rows x tile_d0 elems) from a
// row-major tensor (tensor_d1 x tensor_d0, row stride stride0 elems)
// into LDS at lds_addr, rows packed back-to-back.
// D# layout per CDNA5 ISA 8.3/8.4 (group0 128b, group1 256b).
// ---------------------------------------------------------------------
__device__ __forceinline__ void tdm_load_2d(uint32_t lds_addr, const void* gptr,
                                            uint32_t tensor_d0, uint32_t tensor_d1,
                                            uint32_t tile_d0, uint32_t tile_d1,
                                            uint64_t stride0)
{
    uint64_t ga = (uint64_t)(uintptr_t)gptr;
    u32x4 g0;
    g0[0] = 1u;                                             // count=1 (valid), user mode
    g0[1] = lds_addr;                                       // LDS byte address
    g0[2] = (uint32_t)(ga & 0xFFFFFFFFu);                   // global_addr[31:0]
    g0[3] = (uint32_t)((ga >> 32) & 0x01FFFFFFu) | (2u << 30); // addr[56:32] | type=2
    i32x8 g1;
    g1[0] = (int)(1u << 16);                                // data_size=1 (2 bytes)
    g1[1] = (int)((tensor_d0 & 0xFFFFu) << 16);             // tensor_dim0[15:0] @63:48
    g1[2] = (int)((tensor_d0 >> 16) | ((tensor_d1 & 0xFFFFu) << 16));
    g1[3] = (int)((tensor_d1 >> 16) | (tile_d0 << 16));     // tile_dim0 @127:112
    g1[4] = (int)(tile_d1 & 0xFFFFu);                       // tile_dim1 (tile_dim2=0)
    g1[5] = (int)(uint32_t)(stride0 & 0xFFFFFFFFu);         // dim0 stride low
    g1[6] = (int)(uint32_t)((stride0 >> 32) & 0xFFFFu);     // dim0 stride hi
    g1[7] = 0;
    i32x4 z4 = {0, 0, 0, 0};
#if defined(__clang_major__) && __clang_major__ >= 23
    i32x8 z8 = {0, 0, 0, 0, 0, 0, 0, 0};
    __builtin_amdgcn_tensor_load_to_lds(g0, g1, z4, z4, z8, 0);
#else
    __builtin_amdgcn_tensor_load_to_lds(g0, g1, z4, z4, 0);
#endif
}

// =====================================================================
// GEMM:  C[M,N] = A[M,K] (f32) @ Bt[N,K]^T (bf16, pre-transposed) + bias
// Block = 256 threads = 8 waves; tile 128(M) x 128(N), K step 32.
// A staged with f32->bf16 convert; B tile streamed by the TDM.
// wave w: rows mi=(w&3)*32 (2 subtiles), cols ng=(w>>2)*64 (4 subtiles)
// =====================================================================
__global__ __launch_bounds__(256)
void gemm_bias_bf16wmma(const float* __restrict__ A, const uint16_t* __restrict__ Bt,
                        const float* __restrict__ bias, float* __restrict__ C,
                        int M, int N, int K)
{
    constexpr int BM = 128, BN = 128, BK = 32;
    __shared__ uint16_t sA[BM][BK];   // [m][k] bf16
    __shared__ uint16_t sB[BN][BK];   // [n][k] bf16

    const int tid  = threadIdx.x;
    const int wave = tid >> 5;
    const int lane = tid & 31;
    const int bm = blockIdx.y * BM;
    const int bn = blockIdx.x * BN;
    const int mi = (wave & 3) * 32;
    const int ng = (wave >> 2) * 64;
    const int m  = lane & 15;
    const int hi = lane >> 4;

    v8f acc[2][4];
    #pragma unroll
    for (int a = 0; a < 2; ++a)
        #pragma unroll
        for (int c = 0; c < 4; ++c) acc[a][c] = (v8f){};

    const uint32_t ldsB = (uint32_t)(uintptr_t)(void*)&sB[0][0];

    for (int k0 = 0; k0 < K; k0 += BK) {
        __syncthreads();
        // async: whole 128x32 bf16 B tile via Tensor Data Mover (one wave)
        if (wave == 0)
            tdm_load_2d(ldsB, Bt + (size_t)bn * K + k0,
                        (uint32_t)K, (uint32_t)N, BK, BN, (uint64_t)K);
        // stage A tile (128x32 f32 -> bf16) with all threads
        for (int i = tid; i < BM * BK; i += 256) {
            int mm = i >> 5, kk = i & 31;
            sA[mm][kk] = f2bf(A[(size_t)(bm + mm) * K + (k0 + kk)]);
        }
        if (wave == 0) __builtin_amdgcn_s_wait_tensorcnt(0);
        __syncthreads();

        // A fragments: lane m, VGPR v -> K = (v<4?0:16) + 2*(v%4) + 8*hi
        Frag a0, a1;
        #pragma unroll
        for (int v = 0; v < 8; ++v) {
            int kk = ((v & 4) << 2) + ((v & 3) << 1) + (hi << 3);
            a0.u[v] = *(const uint32_t*)&sA[mi +      m][kk];
            a1.u[v] = *(const uint32_t*)&sA[mi + 16 + m][kk];
        }
        // 4 B fragments: lane n, VGPR v -> K = (v<4?0:16) + 2*(v%4) + 8*hi
        #pragma unroll
        for (int ns = 0; ns < 4; ++ns) {
            Frag bf;
            #pragma unroll
            for (int v = 0; v < 8; ++v) {
                int kk = ((v & 4) << 2) + ((v & 3) << 1) + (hi << 3);
                bf.u[v] = *(const uint32_t*)&sB[ng + ns * 16 + m][kk];
            }
            acc[0][ns] = wmma_bf16(a0, bf, acc[0][ns]);
            acc[1][ns] = wmma_bf16(a1, bf, acc[1][ns]);
        }
    }

    // epilogue: C layout -> VGPR r = row r + 8*hi, col = lane%16
    #pragma unroll
    for (int ms = 0; ms < 2; ++ms) {
        const int row0 = bm + mi + ms * 16 + hi * 8;
        #pragma unroll
        for (int ns = 0; ns < 4; ++ns) {
            const int col = bn + ng + ns * 16 + m;
            const float bia = bias[col];
            #pragma unroll
            for (int r = 0; r < 8; ++r)
                C[(size_t)(row0 + r) * N + col] = acc[ms][ns][r] + bia;
        }
    }
}

// =====================================================================
// Weight prep: W[K][N] f32  ->  Wt[N][K] bf16 (tiled transpose)
// =====================================================================
__global__ __launch_bounds__(256)
void transpose_f32_to_bf16(const float* __restrict__ src, uint16_t* __restrict__ dst,
                           int K, int N)
{
    __shared__ float tile[32][33];
    const int kb = blockIdx.y * 32, nb = blockIdx.x * 32;
    const int tx = threadIdx.x & 31, ty = threadIdx.x >> 5;  // ty 0..7
    #pragma unroll
    for (int r = ty; r < 32; r += 8)
        tile[r][tx] = src[(size_t)(kb + r) * N + (nb + tx)];
    __syncthreads();
    #pragma unroll
    for (int r = ty; r < 32; r += 8)
        dst[(size_t)(nb + r) * K + (kb + tx)] = f2bf(tile[tx][r]);
}

// =====================================================================
// RoPE + f32->bf16.  src rows of 128; seq position s = (row/seqDiv) % S.
// =====================================================================
__global__ __launch_bounds__(256)
void rope_to_bf16(const float* __restrict__ src, uint16_t* __restrict__ dst,
                  int nRows, int seqDiv)
{
    int idx = blockIdx.x * blockDim.x + threadIdx.x;
    if (idx >= nRows * 64) return;
    int row = idx >> 6;
    int j   = idx & 63;
    int s   = (row / seqDiv) % S;
    float inv = powf(10000.0f, -(float)j / 64.0f);
    float ang = (float)s * inv;
    float c = cosf(ang), sn = sinf(ang);
    const float* r = src + (size_t)row * 128;
    float x1 = r[j], x2 = r[j + 64];
    dst[(size_t)row * 128 + j]      = f2bf(x1 * c  - x2 * sn);
    dst[(size_t)row * 128 + j + 64] = f2bf(x1 * sn + x2 * c);
}

__global__ __launch_bounds__(256)
void cvt_f32_bf16(const float* __restrict__ src, uint16_t* __restrict__ dst, int n)
{
    int i = blockIdx.x * blockDim.x + threadIdx.x;
    if (i < n) dst[i] = f2bf(src[i]);
}

// =====================================================================
// MQA flash attention.
// grid = (S/16, B).  block = 512 threads = 16 waves; wave w == head w.
// K tile streamed by TDM (MQA: shared by all 16 heads). Per 32-key block:
// QK^T (8 wmma), online softmax, P.V (8 wmma).
// =====================================================================
__global__ __launch_bounds__(512)
void mqa_attention(const uint16_t* __restrict__ Qb, const uint16_t* __restrict__ Kb,
                   const uint16_t* __restrict__ Vb, const unsigned char* __restrict__ mask,
                   float* __restrict__ attnOut)
{
    constexpr int KB = 32;               // keys per iteration
    __shared__ uint16_t sK [KB][DK];     // [key][d]   8 KB  (TDM dest)
    __shared__ uint16_t sVt[DV][KB];     // [d][key]   8 KB
    __shared__ uint16_t sP [16][16][KB]; // per-wave probs [wave][m][k]  16 KB

    const int tid  = threadIdx.x;
    const int wave = tid >> 5;           // head
    const int lane = tid & 31;
    const int h    = wave;
    const int qt   = blockIdx.x;
    const int b    = blockIdx.y;
    const int qbase = qt * 16;
    const int m  = lane & 15;
    const int hi = lane >> 4;
    const uint32_t ldsK = (uint32_t)(uintptr_t)(void*)&sK[0][0];

    // preload Q A-fragments (4 chunks of K=32) straight from global
    Frag qf[4];
    const uint16_t* qrow = Qb + (((size_t)(b * S + qbase + m)) * H + h) * DK;
    #pragma unroll
    for (int c = 0; c < 4; ++c)
        #pragma unroll
        for (int v = 0; v < 8; ++v) {
            int kk = c * 32 + ((v & 4) << 2) + ((v & 3) << 1) + (hi << 3);
            qf[c].u[v] = *(const uint32_t*)(qrow + kk);
        }

    v8f o[8];
    #pragma unroll
    for (int nt = 0; nt < 8; ++nt) o[nt] = (v8f){};
    float mstat[8], lstat[8];
    #pragma unroll
    for (int r = 0; r < 8; ++r) { mstat[r] = -3.0e38f; lstat[r] = 0.0f; }

    const int nkb = (qbase + 16 + KB - 1) / KB;   // causal key-block bound

    for (int kb = 0; kb < nkb; ++kb) {
        const int k0 = kb * KB;
        __syncthreads();
        // K tile: one TDM op for the whole 32x128 bf16 tile
        if (wave == 0)
            tdm_load_2d(ldsK, Kb + ((size_t)b * S + k0) * DK,
                        DK, BS, DK, KB, DK);
        // V tile: transpose-stage [d][key] with plain loads
        for (int i = tid; i < KB * DV; i += 512) {
            int kk = i >> 7, d = i & 127;
            sVt[d][kk] = Vb[((size_t)b * S + k0 + kk) * DV + d];
        }
        if (wave == 0) __builtin_amdgcn_s_wait_tensorcnt(0);
        __syncthreads();

        // ---- scores: two 16-key subtiles, accumulate over 4 d-chunks ----
        v8f c0 = {}, c1 = {};
        #pragma unroll
        for (int c = 0; c < 4; ++c) {
            Frag b0, b1;
            #pragma unroll
            for (int v = 0; v < 8; ++v) {
                int kk = c * 32 + ((v & 4) << 2) + ((v & 3) << 1) + (hi << 3);
                b0.u[v] = *(const uint32_t*)&sK[m][kk];
                b1.u[v] = *(const uint32_t*)&sK[16 + m][kk];
            }
            c0 = wmma_bf16(qf[c], b0, c0);
            c1 = wmma_bf16(qf[c], b1, c1);
        }

        // ---- scale + causal/pad mask (lane holds col n=m, rows r+8*hi) ----
        const int keyA = k0 + m, keyB = k0 + 16 + m;
        const bool padA = mask[(size_t)b * S + keyA] != 0;
        const bool padB = mask[(size_t)b * S + keyB] != 0;
        #pragma unroll
        for (int r = 0; r < 8; ++r) {
            int qr = qbase + r + 8 * hi;
            float a0 = c0[r] * SCALE; if (padA || keyA > qr) a0 = -3.0e38f;
            float a1 = c1[r] * SCALE; if (padB || keyB > qr) a1 = -3.0e38f;
            c0[r] = a0; c1[r] = a1;
        }

        // ---- online softmax (reduce across the 16-lane half) ----
        #pragma unroll
        for (int r = 0; r < 8; ++r) {
            float v = fmaxf(c0[r], c1[r]);
            v = fmaxf(v, __shfl_xor(v, 1, 32));
            v = fmaxf(v, __shfl_xor(v, 2, 32));
            v = fmaxf(v, __shfl_xor(v, 4, 32));
            v = fmaxf(v, __shfl_xor(v, 8, 32));
            float nm = fmaxf(mstat[r], v);
            float p0 = __expf(c0[r] - nm);
            float p1 = __expf(c1[r] - nm);
            float rs = p0 + p1;
            rs += __shfl_xor(rs, 1, 32);
            rs += __shfl_xor(rs, 2, 32);
            rs += __shfl_xor(rs, 4, 32);
            rs += __shfl_xor(rs, 8, 32);
            float corr = __expf(mstat[r] - nm);
            lstat[r] = lstat[r] * corr + rs;
            mstat[r] = nm;
            c0[r] = p0; c1[r] = p1;
            #pragma unroll
            for (int nt = 0; nt < 8; ++nt) o[nt][r] *= corr;
        }

        // ---- re-layout probs C->A through per-wave LDS tile ----
        #pragma unroll
        for (int r = 0; r < 8; ++r) {
            sP[wave][r + 8 * hi][m]      = f2bf(c0[r]);
            sP[wave][r + 8 * hi][16 + m] = f2bf(c1[r]);
        }
        __syncthreads();
        Frag pf;
        #pragma unroll
        for (int v = 0; v < 8; ++v) {
            int kk = ((v & 4) << 2) + ((v & 3) << 1) + (hi << 3);
            pf.u[v] = *(const uint32_t*)&sP[wave][m][kk];
        }

        // ---- P @ V : 8 output subtiles of 16 dims ----
        #pragma unroll
        for (int nt = 0; nt < 8; ++nt) {
            Frag bf;
            #pragma unroll
            for (int v = 0; v < 8; ++v) {
                int kk = ((v & 4) << 2) + ((v & 3) << 1) + (hi << 3);
                bf.u[v] = *(const uint32_t*)&sVt[nt * 16 + m][kk];
            }
            o[nt] = wmma_bf16(pf, bf, o[nt]);
        }
    }

    // epilogue: normalize, write attn (B,S,H*DV) f32
    #pragma unroll
    for (int r = 0; r < 8; ++r) {
        float inv = 1.0f / lstat[r];
        int srow = qbase + r + 8 * hi;
        float* orow = attnOut + ((size_t)(b * S + srow)) * (H * DV) + h * DV;
        #pragma unroll
        for (int nt = 0; nt < 8; ++nt)
            orow[nt * 16 + m] = o[nt][r] * inv;
    }
}

// =====================================================================
// Launcher
// =====================================================================
extern "C" void kernel_launch(void* const* d_in, const int* in_sizes, int n_in,
                              void* d_out, int out_size, void* d_ws, size_t ws_size,
                              hipStream_t stream) {
    (void)in_sizes; (void)n_in; (void)out_size; (void)ws_size;
    const float* q_in = (const float*)d_in[0];
    const float* k_in = (const float*)d_in[1];
    const float* v_in = (const float*)d_in[2];
    const unsigned char* mask = (const unsigned char*)d_in[3];
    const float* Wq = (const float*)d_in[4];
    const float* bq = (const float*)d_in[5];
    const float* Wk = (const float*)d_in[6];
    const float* bk = (const float*)d_in[7];
    const float* Wv = (const float*)d_in[8];
    const float* bv = (const float*)d_in[9];
    const float* Wo = (const float*)d_in[10];
    const float* bo = (const float*)d_in[11];
    float* out = (float*)d_out;

    // workspace carve-up (~75 MB)
    char* w = (char*)d_ws;
    float*    qf32 = (float*)w;    w += (size_t)BS * D   * 4;   // 33.5 MB
    float*    kf32 = (float*)w;    w += (size_t)BS * DK  * 4;   //  2.1 MB
    float*    vf32 = (float*)w;    w += (size_t)BS * DV  * 4;   //  2.1 MB
    uint16_t* qbf  = (uint16_t*)w; w += (size_t)BS * D   * 2;   // 16.8 MB
    uint16_t* kbf  = (uint16_t*)w; w += (size_t)BS * DK  * 2;   //  1.0 MB
    uint16_t* vbf  = (uint16_t*)w; w += (size_t)BS * DV  * 2;   //  1.0 MB
    uint16_t* WqT  = (uint16_t*)w; w += (size_t)D  * D   * 2;   //  8.4 MB
    uint16_t* WkT  = (uint16_t*)w; w += (size_t)DK * D   * 2;   //  0.5 MB
    uint16_t* WvT  = (uint16_t*)w; w += (size_t)DV * D   * 2;   //  0.5 MB
    uint16_t* WoT  = (uint16_t*)w; w += (size_t)D  * D   * 2;   //  8.4 MB
    float*    attnf = qf32;  // qf32 dead after RoPE -> reuse for attention out

    // 0) weight prep: transpose + bf16 ([K][N] -> [N][K])
    transpose_f32_to_bf16<<<dim3(D  / 32, D / 32), 256, 0, stream>>>(Wq, WqT, D, D);
    transpose_f32_to_bf16<<<dim3(DK / 32, D / 32), 256, 0, stream>>>(Wk, WkT, D, DK);
    transpose_f32_to_bf16<<<dim3(DV / 32, D / 32), 256, 0, stream>>>(Wv, WvT, D, DV);
    transpose_f32_to_bf16<<<dim3(D  / 32, D / 32), 256, 0, stream>>>(Wo, WoT, D, D);

    // 1) projections (WMMA GEMMs, B tiles via TDM)
    gemm_bias_bf16wmma<<<dim3(D  / 128, BS / 128), 256, 0, stream>>>(q_in, WqT, bq, qf32, BS, D,  D);
    gemm_bias_bf16wmma<<<dim3(DK / 128, BS / 128), 256, 0, stream>>>(k_in, WkT, bk, kf32, BS, DK, D);
    gemm_bias_bf16wmma<<<dim3(DV / 128, BS / 128), 256, 0, stream>>>(v_in, WvT, bv, vf32, BS, DV, D);

    // 2) RoPE (Q per head, K per row) + bf16 convert; V straight convert
    {
        int nRowsQ = BS * H;                     // 65536 head-rows of 128
        rope_to_bf16<<<(nRowsQ * 64 + 255) / 256, 256, 0, stream>>>(qf32, qbf, nRowsQ, H);
        int nRowsK = BS;
        rope_to_bf16<<<(nRowsK * 64 + 255) / 256, 256, 0, stream>>>(kf32, kbf, nRowsK, 1);
        int nV = BS * DV;
        cvt_f32_bf16<<<(nV + 255) / 256, 256, 0, stream>>>(vf32, vbf, nV);
    }

    // 3) flash attention: one block per (16-query tile, batch); 16 waves = 16 heads
    mqa_attention<<<dim3(S / 16, Bb), 512, 0, stream>>>(qbf, kbf, vbf, mask, attnf);

    // 4) output projection
    gemm_bias_bf16wmma<<<dim3(D / 128, BS / 128), 256, 0, stream>>>(attnf, WoT, bo, out, BS, D, D);
}
